// MaskedAttentionEmb_31284541784113
// MI455X (gfx1250) — matrix operand order repaired and checked by reading, not compiled, not run on hardware
//
#include <hip/hip_runtime.h>
#include <math.h>

// ---------------------------------------------------------------------------
// MI455X (gfx1250). GEMMs on v_wmma_f32_16x16x32_f16 with f32 accumulation.
// Weights are staged f32->f16 once per call; activation producers (LN, attn,
// GELU epilogues) emit f16 copies, so the WMMA inner loop is pure
// global_load_b128 + v_wmma (no converts, half the operand bytes).
// ---------------------------------------------------------------------------

typedef __attribute__((ext_vector_type(16))) _Float16 v16h;
typedef __attribute__((ext_vector_type(8)))  _Float16 v8h;
typedef __attribute__((ext_vector_type(4)))  _Float16 v4h;
typedef __attribute__((ext_vector_type(8)))  float    v8f;

#define K_S      1024
#define K_B      4
#define K_EMB    128
#define K_HID    768
#define K_LAYERS 4
#define K_CW     3
#define K_NHEAD  12
#define K_HDIM   64
#define K_T      (K_S - 4 * K_CW)   /* 1012 */
#define K_M      (K_S * K_B)        /* 4096 token rows */
#define K_G3     (3 * K_HID)        /* 2304 */
#define K_FF     (4 * K_HID)        /* 3072 */
#define K_EPS    1e-5f

__device__ __forceinline__ float gelu_exact(float x) {
  return 0.5f * x * (1.0f + erff(x * 0.70710678118654752440f));
}

__device__ __forceinline__ float softplus_stable(float x) {
  return fmaxf(x, 0.0f) + log1pf(expf(-fabsf(x)));
}

// sinusoidal positional encoding, matching the reference exactly
__device__ __forceinline__ float pos_enc(int s, int n) {
  const float c = -9.210340371976184f / (float)K_HID;  // -ln(10000)/HID
  float div = expf((float)(n & ~1) * c);
  float ang = (float)s * div;
  return (n & 1) ? cosf(ang) : sinf(ang);
}

// ---------------------------------------------------------------------------
// f32 -> f16 staging (weights). n must be a multiple of 8 (true for all uses).
// ---------------------------------------------------------------------------
__global__ __launch_bounds__(256) void cvt_f16_kernel(
    const float* __restrict__ in, _Float16* __restrict__ out, int n) {
  int i = (int)(blockIdx.x * blockDim.x + threadIdx.x) * 8;
  if (i >= n) return;
  float4 a = *reinterpret_cast<const float4*>(in + i);
  float4 b = *reinterpret_cast<const float4*>(in + i + 4);
  v8h o;
  o[0] = (_Float16)a.x; o[1] = (_Float16)a.y;
  o[2] = (_Float16)a.z; o[3] = (_Float16)a.w;
  o[4] = (_Float16)b.x; o[5] = (_Float16)b.y;
  o[6] = (_Float16)b.z; o[7] = (_Float16)b.w;
  *reinterpret_cast<v8h*>(out + i) = o;
}

// ---------------------------------------------------------------------------
// WMMA fragment loads (wave32), operands already f16; contiguous k order shared by A and B, not the ISA table order.
// A (16x32, MxK): lane l -> M=l&15, half=l>>4:
//   v[0..7] pack K = kb+half*16+{0..15}
// B (32x16, KxN): lane l -> N=l&15, half=l>>4: v[0..7] pack K = kb+half*16+{0..15}
// ---------------------------------------------------------------------------
__device__ __forceinline__ v16h load_frag_a_h(const _Float16* __restrict__ rowp,
                                              int kb, int half) {
  return *reinterpret_cast<const v16h*>(rowp + kb + half * 16);
}

__device__ __forceinline__ v16h load_frag_b_h(const _Float16* __restrict__ rowp,
                                              int kb, int half) {
  return *reinterpret_cast<const v16h*>(rowp + kb + half * 16);
}

// ---------------------------------------------------------------------------
// GEMM: act(A[M,K] * W[N,K]^T + bias) -> optional f32 out Cf and/or f16 out Ch.
// One wave computes a 32x64 macro tile (2x4 WMMA sub-tiles), K steps of 32.
// M % 32 == 0, N % 64 == 0, K % 32 == 0 for every call made here.
// ---------------------------------------------------------------------------
#define GBM 2
#define GBN 4
__global__ __launch_bounds__(256) void wmma_gemm_kernel(
    const _Float16* __restrict__ A, const _Float16* __restrict__ W,
    const float* __restrict__ bias, float* __restrict__ Cf,
    _Float16* __restrict__ Ch, int M, int N, int K, int act, int addPE) {
  int wave = (int)((blockIdx.x * blockDim.x + threadIdx.x) >> 5);
  int lane = (int)(threadIdx.x & 31);
  int mt = M >> 5;   // macro tiles along M
  int nt = N >> 6;   // macro tiles along N
  if (wave >= mt * nt) return;
  int tm = wave % mt, tn = wave / mt;
  int m0 = tm << 5, n0 = tn << 6;
  int half = lane >> 4, lq = lane & 15;

  const v8f zerov = {0.f, 0.f, 0.f, 0.f, 0.f, 0.f, 0.f, 0.f};
  v8f acc[GBM][GBN];
#pragma unroll
  for (int i = 0; i < GBM; ++i)
#pragma unroll
    for (int j = 0; j < GBN; ++j) acc[i][j] = zerov;

  const _Float16* arow[GBM];
#pragma unroll
  for (int i = 0; i < GBM; ++i)
    arow[i] = A + (size_t)(m0 + i * 16 + lq) * (size_t)K;
  const _Float16* brow[GBN];
#pragma unroll
  for (int j = 0; j < GBN; ++j)
    brow[j] = W + (size_t)(n0 + j * 16 + lq) * (size_t)K;

  for (int kb = 0; kb < K; kb += 32) {
    if (kb + 32 < K) {  // pull the next K slab toward the caches
      __builtin_prefetch(arow[0] + kb + 32, 0, 3);
      __builtin_prefetch(brow[0] + kb + 32, 0, 3);
    }
    v16h af[GBM];
#pragma unroll
    for (int i = 0; i < GBM; ++i) af[i] = load_frag_a_h(arow[i], kb, half);
    v16h bf[GBN];
#pragma unroll
    for (int j = 0; j < GBN; ++j) bf[j] = load_frag_b_h(brow[j], kb, half);
#pragma unroll
    for (int i = 0; i < GBM; ++i)
#pragma unroll
      for (int j = 0; j < GBN; ++j)
        acc[i][j] = __builtin_amdgcn_wmma_f32_16x16x32_f16(
            false, af[i], false, bf[j], (short)0, acc[i][j], false, false);
  }

  // Epilogue. C/D layout: lane l -> N = l&15; VGPR e -> M = e + 8*(l>>4).
#pragma unroll
  for (int i = 0; i < GBM; ++i) {
#pragma unroll
    for (int j = 0; j < GBN; ++j) {
      int n = n0 + j * 16 + lq;
      float bv = bias[n];
#pragma unroll
      for (int e = 0; e < 8; ++e) {
        int m = m0 + i * 16 + e + 8 * half;
        float v = acc[i][j][e] + bv;
        if (act == 1) v = gelu_exact(v);
        if (addPE) v += pos_enc(m / K_B, n);
        size_t idx = (size_t)m * (size_t)N + (size_t)n;
        if (Cf) Cf[idx] = v;
        if (Ch) Ch[idx] = (_Float16)v;
      }
    }
  }
}

// ---------------------------------------------------------------------------
// Embedding gather + LayerNorm -> z (f32, for loss) and zh (f16, for proj).
// One wave per row (EMB = 128 -> 4 elements per lane).
// ---------------------------------------------------------------------------
__global__ __launch_bounds__(256) void emb_ln_kernel(
    const int* __restrict__ seq, const float* __restrict__ emb,
    const float* __restrict__ g, const float* __restrict__ bb,
    float* __restrict__ z, _Float16* __restrict__ zh) {
  int wave = (int)((blockIdx.x * blockDim.x + threadIdx.x) >> 5);
  int lane = (int)(threadIdx.x & 31);
  if (wave >= K_M) return;
  int s = wave / K_B, b = wave % K_B;
  int id = seq[b * K_S + s];
  const float* e = emb + (size_t)id * K_EMB;
  float sm = 0.f, sq = 0.f;
#pragma unroll
  for (int i = lane; i < K_EMB; i += 32) {
    float v = e[i];
    sm += v; sq += v * v;
  }
#pragma unroll
  for (int off = 16; off > 0; off >>= 1) {
    sm += __shfl_xor(sm, off, 32);
    sq += __shfl_xor(sq, off, 32);
  }
  float mean = sm * (1.0f / K_EMB);
  float var = sq * (1.0f / K_EMB) - mean * mean;
  float rstd = rsqrtf(var + K_EPS);
  float* zr = z + (size_t)wave * K_EMB;
  _Float16* zrh = zh + (size_t)wave * K_EMB;
#pragma unroll
  for (int i = lane; i < K_EMB; i += 32) {
    float v = (e[i] - mean) * rstd * g[i] + bb[i];
    zr[i] = v;
    zrh[i] = (_Float16)v;
  }
}

// ---------------------------------------------------------------------------
// Residual-add + LayerNorm, one wave per row. y / outh may be null.
// In-place (out == x) is safe: each element read+written by its own lane.
// ---------------------------------------------------------------------------
__global__ __launch_bounds__(256) void add_ln_kernel(
    const float* __restrict__ x, const float* __restrict__ y,
    const float* __restrict__ g, const float* __restrict__ bb,
    float* __restrict__ out, _Float16* __restrict__ outh, int rows, int H) {
  int wave = (int)((blockIdx.x * blockDim.x + threadIdx.x) >> 5);
  int lane = (int)(threadIdx.x & 31);
  if (wave >= rows) return;
  const float* xr = x + (size_t)wave * H;
  const float* yr = y ? (y + (size_t)wave * H) : nullptr;
  float sm = 0.f, sq = 0.f;
  for (int i = lane; i < H; i += 32) {
    float v = xr[i] + (yr ? yr[i] : 0.f);
    sm += v; sq += v * v;
  }
#pragma unroll
  for (int off = 16; off > 0; off >>= 1) {
    sm += __shfl_xor(sm, off, 32);
    sq += __shfl_xor(sq, off, 32);
  }
  float mean = sm / (float)H;
  float var = sq / (float)H - mean * mean;
  float rstd = rsqrtf(var + K_EPS);
  float* orow = out + (size_t)wave * H;
  _Float16* ohrow = outh ? (outh + (size_t)wave * H) : nullptr;
  for (int i = lane; i < H; i += 32) {
    float v = xr[i] + (yr ? yr[i] : 0.f);
    float r = (v - mean) * rstd * g[i] + bb[i];
    orow[i] = r;
    if (ohrow) ohrow[i] = (_Float16)r;
  }
}

// ---------------------------------------------------------------------------
// Banded attention: gaps g in [0,12], g != 6 (12 taps), softmax, V-blend.
// One thread per (s, b, head). Output is f16-only (feeds the WO GEMM).
// ---------------------------------------------------------------------------
__global__ __launch_bounds__(256) void attn_kernel(
    const float* __restrict__ qkv, _Float16* __restrict__ outh) {
  int t = (int)(blockIdx.x * blockDim.x + threadIdx.x);
  if (t >= K_S * K_B * K_NHEAD) return;
  int h = t % K_NHEAD;
  int tmp = t / K_NHEAD;
  int b = tmp % K_B;
  int s = tmp / K_B;
  const float* q = qkv + (size_t)(s * K_B + b) * K_G3 + h * K_HDIM;

  float sc[12];
#pragma unroll
  for (int g = 0; g <= 12; ++g) {
    if (g == 6) continue;
    int slot = (g < 6) ? g : g - 1;
    int j = s + g;
    float d = -1e30f;
    if (j < K_S) {
      const float* kp = qkv + (size_t)(j * K_B + b) * K_G3 + K_HID + h * K_HDIM;
      float acc = 0.f;
#pragma unroll
      for (int i = 0; i < K_HDIM; i += 4) {
        float4 qa = *reinterpret_cast<const float4*>(q + i);
        float4 ka = *reinterpret_cast<const float4*>(kp + i);
        acc += qa.x * ka.x + qa.y * ka.y + qa.z * ka.z + qa.w * ka.w;
      }
      d = acc * 0.125f;  // 1/sqrt(64)
    }
    sc[slot] = d;
  }
  float mx = sc[0];
#pragma unroll
  for (int i = 1; i < 12; ++i) mx = fmaxf(mx, sc[i]);
  float sum = 0.f;
#pragma unroll
  for (int i = 0; i < 12; ++i) { sc[i] = expf(sc[i] - mx); sum += sc[i]; }
  float inv = 1.0f / sum;
#pragma unroll
  for (int i = 0; i < 12; ++i) sc[i] *= inv;

  _Float16* o = outh + (size_t)(s * K_B + b) * K_HID + h * K_HDIM;
#pragma unroll
  for (int dc = 0; dc < K_HDIM; dc += 4) {
    float4 a; a.x = 0.f; a.y = 0.f; a.z = 0.f; a.w = 0.f;
#pragma unroll
    for (int i = 0; i < 12; ++i) {
      int g = (i < 6) ? i : i + 1;
      int j = s + g;
      if (j < K_S) {
        const float* vp =
            qkv + (size_t)(j * K_B + b) * K_G3 + 2 * K_HID + h * K_HDIM + dc;
        float4 va = *reinterpret_cast<const float4*>(vp);
        float w = sc[i];
        a.x += w * va.x; a.y += w * va.y; a.z += w * va.z; a.w += w * va.w;
      }
    }
    v4h oh;
    oh[0] = (_Float16)a.x; oh[1] = (_Float16)a.y;
    oh[2] = (_Float16)a.z; oh[3] = (_Float16)a.w;
    *reinterpret_cast<v4h*>(o + dc) = oh;
  }
}

// ---------------------------------------------------------------------------
// Loss: per (t, b) compute softplus(-pos) + softplus(neg); block partials.
// ---------------------------------------------------------------------------
__global__ __launch_bounds__(256) void loss_kernel(
    const float* __restrict__ z, const float* __restrict__ zout,
    const float* __restrict__ emb, const int* __restrict__ neg_ids,
    const float* __restrict__ eg, const float* __restrict__ eb,
    float* __restrict__ partials) {
  int tid = (int)(blockIdx.x * blockDim.x + threadIdx.x);
  float contrib = 0.f;
  if (tid < K_T * K_B) {
    int t = tid / K_B, b = tid % K_B;
    int r = (t + K_CW + 1) * K_B + b;  // target row
    const float4* zr = reinterpret_cast<const float4*>(z + (size_t)r * K_EMB);
    const float4* zo = reinterpret_cast<const float4*>(zout + (size_t)r * K_EMB);
    float pos = 0.f;
#pragma unroll
    for (int i = 0; i < K_EMB / 4; ++i) {
      float4 a = zr[i], c = zo[i];
      pos += a.x * c.x + a.y * c.y + a.z * c.z + a.w * c.w;
    }
    int id = neg_ids[t * K_B + b];
    const float4* er = reinterpret_cast<const float4*>(emb + (size_t)id * K_EMB);
    float sm = 0.f, sq = 0.f;
#pragma unroll
    for (int i = 0; i < K_EMB / 4; ++i) {
      float4 e = er[i];
      sm += e.x + e.y + e.z + e.w;
      sq += e.x * e.x + e.y * e.y + e.z * e.z + e.w * e.w;
    }
    float mean = sm * (1.0f / K_EMB);
    float var = sq * (1.0f / K_EMB) - mean * mean;
    float rstd = rsqrtf(var + K_EPS);
    const float4* gg = reinterpret_cast<const float4*>(eg);
    const float4* bb = reinterpret_cast<const float4*>(eb);
    float neg = 0.f;
#pragma unroll
    for (int i = 0; i < K_EMB / 4; ++i) {
      float4 e = er[i], c = zo[i], g4 = gg[i], b4 = bb[i];
      neg += ((e.x - mean) * rstd * g4.x + b4.x) * c.x;
      neg += ((e.y - mean) * rstd * g4.y + b4.y) * c.y;
      neg += ((e.z - mean) * rstd * g4.z + b4.z) * c.z;
      neg += ((e.w - mean) * rstd * g4.w + b4.w) * c.w;
    }
    contrib = softplus_stable(-pos) + softplus_stable(neg);
  }
  __shared__ float red[256];
  red[threadIdx.x] = contrib;
  __syncthreads();
#pragma unroll
  for (int st = 128; st > 0; st >>= 1) {
    if ((int)threadIdx.x < st) red[threadIdx.x] += red[threadIdx.x + st];
    __syncthreads();
  }
  if (threadIdx.x == 0) partials[blockIdx.x] = red[0];
}

__global__ void finalize_kernel(const float* __restrict__ partials, int n,
                                float* __restrict__ outp, float invdenom) {
  if (blockIdx.x == 0 && threadIdx.x == 0) {
    float s = 0.f;
    for (int i = 0; i < n; ++i) s += partials[i];  // fixed order: deterministic
    outp[0] = s * invdenom;
  }
}

// ---------------------------------------------------------------------------
// Host-side orchestration (all launches on `stream`, graph-capture safe).
// ---------------------------------------------------------------------------
extern "C" void kernel_launch(void* const* d_in, const int* in_sizes, int n_in,
                              void* d_out, int out_size, void* d_ws,
                              size_t ws_size, hipStream_t stream) {
  (void)in_sizes; (void)n_in; (void)out_size; (void)ws_size;
  const int*   seq     = (const int*)d_in[0];
  const int*   neg_ids = (const int*)d_in[1];
  const float* emb     = (const float*)d_in[2];
  const float* emb_g   = (const float*)d_in[3];
  const float* emb_b   = (const float*)d_in[4];
  const float* proj_w  = (const float*)d_in[5];
  const float* proj_b  = (const float*)d_in[6];
  const float* wqkv    = (const float*)d_in[7];
  const float* bqkv    = (const float*)d_in[8];
  const float* wo      = (const float*)d_in[9];
  const float* bo      = (const float*)d_in[10];
  const float* ln1g    = (const float*)d_in[11];
  const float* ln1b    = (const float*)d_in[12];
  const float* w1      = (const float*)d_in[13];
  const float* b1      = (const float*)d_in[14];
  const float* w2      = (const float*)d_in[15];
  const float* b2      = (const float*)d_in[16];
  const float* ln2g    = (const float*)d_in[17];
  const float* ln2b    = (const float*)d_in[18];
  const float* out_w   = (const float*)d_in[19];
  const float* out_b   = (const float*)d_in[20];
  const float* olng    = (const float*)d_in[21];
  const float* olnb    = (const float*)d_in[22];

  // ---- workspace layout: f32 region then f16 region ----
  float* ws   = (float*)d_ws;
  float* z    = ws;                                   // 4096*128
  float* x    = z    + (size_t)K_M * K_EMB;           // 4096*768
  float* qkv  = x    + (size_t)K_M * K_HID;           // 4096*2304
  float* ff2  = qkv  + (size_t)K_M * K_G3;            // 4096*768
  float* partials = ff2 + (size_t)K_M * K_HID;        // 16
  float* attnproj = qkv;  // qkv region is free once attn is consumed
  float* zout     = qkv;  // and again after the layer stack

  _Float16* h = (_Float16*)(partials + 32);
  _Float16* zh    = h;                                 // 4096*128
  _Float16* xh    = zh    + (size_t)K_M * K_EMB;       // 4096*768
  _Float16* attnh = xh    + (size_t)K_M * K_HID;       // 4096*768
  _Float16* ff1h  = attnh + (size_t)K_M * K_HID;       // 4096*3072
  _Float16* pwh   = ff1h  + (size_t)K_M * K_FF;        // 768*128
  _Float16* qkvwh = pwh   + (size_t)K_HID * K_EMB;     // 4*2304*768
  _Float16* wowh  = qkvwh + (size_t)K_LAYERS * K_G3 * K_HID;   // 4*768*768
  _Float16* w1h   = wowh  + (size_t)K_LAYERS * K_HID * K_HID;  // 4*3072*768
  _Float16* w2h   = w1h   + (size_t)K_LAYERS * K_FF * K_HID;   // 4*768*3072
  _Float16* outwh = w2h   + (size_t)K_LAYERS * K_HID * K_FF;   // 128*768

  auto cvt = [&](const float* in, _Float16* out, size_t n) {
    int blocks = (int)((n / 8 + 255) / 256);
    cvt_f16_kernel<<<dim3(blocks), dim3(256), 0, stream>>>(in, out, (int)n);
  };
  auto gemm = [&](const _Float16* A, const _Float16* W, const float* bias,
                  float* Cf, _Float16* Ch, int M, int N, int K, int act,
                  int addPE) {
    int waves = (M >> 5) * (N >> 6);
    int blocks = (waves + 7) / 8;  // 8 waves (256 threads) per block
    wmma_gemm_kernel<<<dim3(blocks), dim3(256), 0, stream>>>(
        A, W, bias, Cf, Ch, M, N, K, act, addPE);
  };
  auto add_ln = [&](const float* xx, const float* yy, const float* g,
                    const float* bb, float* out, _Float16* outh, int rows,
                    int H) {
    int blocks = (rows + 7) / 8;
    add_ln_kernel<<<dim3(blocks), dim3(256), 0, stream>>>(xx, yy, g, bb, out,
                                                          outh, rows, H);
  };

  // 0) stage all weights f32 -> f16 (streamed once: ~113 MB -> ~57 MB)
  cvt(proj_w, pwh, (size_t)K_HID * K_EMB);
  cvt(wqkv, qkvwh, (size_t)K_LAYERS * K_G3 * K_HID);
  cvt(wo, wowh, (size_t)K_LAYERS * K_HID * K_HID);
  cvt(w1, w1h, (size_t)K_LAYERS * K_FF * K_HID);
  cvt(w2, w2h, (size_t)K_LAYERS * K_HID * K_FF);
  cvt(out_w, outwh, (size_t)K_EMB * K_HID);

  // 1) embedding gather + LN -> z (f32) + zh (f16)
  emb_ln_kernel<<<dim3(K_M / 8), dim3(256), 0, stream>>>(seq, emb, emb_g,
                                                         emb_b, z, zh);
  // 2) proj EMB->HID, GELU, +pos-enc -> x (f32 residual) + xh (f16)
  gemm(zh, pwh, proj_b, x, xh, K_M, K_HID, K_EMB, /*gelu*/ 1, /*pe*/ 1);

  // 3) transformer layers
  for (int l = 0; l < K_LAYERS; ++l) {
    gemm(xh, qkvwh + (size_t)l * K_G3 * K_HID, bqkv + (size_t)l * K_G3, qkv,
         nullptr, K_M, K_G3, K_HID, 0, 0);
    attn_kernel<<<dim3(K_S * K_B * K_NHEAD / 256), dim3(256), 0, stream>>>(
        qkv, attnh);
    gemm(attnh, wowh + (size_t)l * K_HID * K_HID, bo + (size_t)l * K_HID,
         attnproj, nullptr, K_M, K_HID, K_HID, 0, 0);
    add_ln(x, attnproj, ln1g + (size_t)l * K_HID, ln1b + (size_t)l * K_HID, x,
           xh, K_M, K_HID);
    gemm(xh, w1h + (size_t)l * K_FF * K_HID, b1 + (size_t)l * K_FF, nullptr,
         ff1h, K_M, K_FF, K_HID, /*gelu*/ 1, 0);
    gemm(ff1h, w2h + (size_t)l * K_HID * K_FF, b2 + (size_t)l * K_HID, ff2,
         nullptr, K_M, K_HID, K_FF, 0, 0);
    add_ln(x, ff2, ln2g + (size_t)l * K_HID, ln2b + (size_t)l * K_HID, x, xh,
           K_M, K_HID);
  }

  // 4) out proj HID->EMB + LN -> zout (f32)
  gemm(xh, outwh, out_b, zout, nullptr, K_M, K_EMB, K_HID, 0, 0);
  add_ln(zout, nullptr, olng, olnb, zout, nullptr, K_M, K_EMB);

  // 5) contrastive BCE loss -> scalar
  const int TB = K_T * K_B;                // 4048
  const int LBLK = (TB + 255) / 256;       // 16
  loss_kernel<<<dim3(LBLK), dim3(256), 0, stream>>>(z, zout, emb, neg_ids,
                                                    emb_g, emb_b, partials);
  finalize_kernel<<<dim3(1), dim3(32), 0, stream>>>(
      partials, LBLK, (float*)d_out, 1.0f / (2.0f * (float)TB));
}
